// HMP_SchNetEnergyCharge_77017353552145
// MI455X (gfx1250) — matrix operand order repaired and checked by reading, not compiled, not run on hardware
//
#include <hip/hip_runtime.h>
#include <hip/hip_bf16.h>
#include <cstdint>

// ---------------- problem constants ----------------
constexpr int cN  = 2048;   // nodes
constexpr int cE  = 65536;  // edges
constexpr int cH  = 128;    // hidden
constexpr int cF  = 128;    // filter
constexpr int cNG = 50;     // gaussians
constexpr int cNGP= 64;     // padded gaussians (K-dim for WMMA)
constexpr int cSD = 32;
constexpr int cMSH= 64;
constexpr int cK  = 512;    // selected nodes
constexpr int cL  = 4;
constexpr int cG  = 16;

#define GSTEP  (10.0f/49.0f)
#define GCOEFF (-0.5f/(GSTEP*GSTEP))
#define PI_F   3.14159265358979f

typedef __attribute__((ext_vector_type(16))) _Float16 v16h;
typedef __attribute__((ext_vector_type(8)))  _Float16 v8h;
typedef __attribute__((ext_vector_type(8)))  float    v8f;

__device__ __forceinline__ float ssp_f(float x){           // softplus(x) - log(2), overflow-safe
  float ax = fabsf(x);
  return fmaxf(x, 0.f) + log1pf(__expf(-ax)) - 0.69314718055994531f;
}
__device__ __forceinline__ float silu_f(float x){ return x * (1.f/(1.f+__expf(-x))); }
__device__ __forceinline__ float sigm_f(float x){ return 1.f/(1.f+__expf(-x)); }

// ---------------- WMMA fragment helpers (CDNA5 16x16x32 f16, ISA 7.12.2) ----------------
// A (16x32 tile): lane L holds row m=L&15; per-lane data = two contiguous 8-half chunks
// at k0 + 8*(L>>4) and k0 + 8*(L>>4) + 16  -> two aligned b128 loads.
__device__ __forceinline__ v16h load_a_frag(const _Float16* A, int lda, int m0, int k0, int lane){
  int m  = m0 + (lane & 15);
  int kh = (lane >> 4) * 8;
  const _Float16* p = A + (size_t)m * lda + k0 + kh;
  v8h lo = *(const v8h*)(p);
  v8h hi = *(const v8h*)(p + 16);
  return __builtin_shufflevector(lo, hi, 0,1,2,3,4,5,6,7,8,9,10,11,12,13,14,15);
}
// B pre-swizzled fragment-major: [ntile][kstep][lane][16 halfs]; one aligned 32B load per lane.
__device__ __forceinline__ v16h load_b_swz(const _Float16* Bs, int Kd, int k0, int n0, int lane){
  size_t idx = ((size_t)(n0 >> 4) * (Kd >> 5) + (k0 >> 5)) * 32 + lane;
  return ((const v16h*)Bs)[idx];
}
#define WMMA_F32(a,b,c) __builtin_amdgcn_wmma_f32_16x16x32_f16(false,(a),false,(b),(short)0,(c),false,false)
// D layout: lane L -> col n = L&15, rows m = r + 8*(L>>4), r=0..7

// ---------------- tiny utility kernels ----------------
__global__ void k_zero(float* p, int n){ int t = blockIdx.x*blockDim.x + threadIdx.x; if (t < n) p[t] = 0.f; }
__global__ void k_f2h(const float* in, _Float16* out, int n){
  int t = blockIdx.x*blockDim.x + threadIdx.x; if (t < n) out[t] = (_Float16)in[t];
}
// Convert f32 weight (rows roff..roff+kvalid-1 of a (rows x Nn) row-major matrix, ld=ldb)
// into swizzled fragment-major f16 B of logical shape (Kd x Nn), zero-padded beyond kvalid.
__global__ void k_swzB(const float* B, int ldb, int roff, int kvalid, int Kd, int Nn, _Float16* out){
  int t = blockIdx.x*blockDim.x + threadIdx.x;
  int total = Nn * Kd;
  if (t >= total) return;
  int e = t & 15; int frag = t >> 4;
  int lane = frag & 31; int tmp = frag >> 5;
  int ksteps = Kd >> 5;
  int kstep = tmp % ksteps; int ntile = tmp / ksteps;
  int n = (ntile << 4) + (lane & 15);
  int k = (kstep << 5) + ((lane >> 4) << 3) + (e & 7) + ((e >> 3) << 4);
  float v = (k < kvalid) ? B[(size_t)(roff + k) * ldb + n] : 0.f;
  out[t] = (_Float16)v;
}

// ---------------- graph setup ----------------
__global__ void k_embed(const int* atoms, const float* emb, float* h){
  int t = blockIdx.x*blockDim.x + threadIdx.x; if (t >= cN*cH) return;
  int v = t >> 7, c = t & 127;
  h[t] = emb[atoms[v]*cH + c];
}
__global__ void k_edge_geom(const int* ei, const float* pos, _Float16* ea_h, float* Cc){
  int e = blockIdx.x*blockDim.x + threadIdx.x; if (e >= cE) return;
  int r = ei[e], c = ei[cE + e];
  float dx = pos[r*3+0]-pos[c*3+0], dy = pos[r*3+1]-pos[c*3+1], dz = pos[r*3+2]-pos[c*3+2];
  float d = sqrtf(dx*dx + dy*dy + dz*dz + 1e-9f);
  Cc[e] = 0.5f * (__cosf(d * (PI_F/10.f)) + 1.f);
  for (int g = 0; g < cNGP; ++g){
    float v = 0.f;
    if (g < cNG){ float t2 = d - GSTEP*g; v = __expf(GCOEFF*t2*t2); }
    ea_h[(size_t)e*cNGP + g] = (_Float16)v;
  }
}

// ---------------- generic WMMA GEMM: C = act(A@B + bias) [+ resid] ----------------
// grid.x = M/16, grid.y = ceil(N/128); block = 256 (8 waves, one 16-col tile each)
template<int ACT>
__global__ void k_gemm(const _Float16* __restrict__ A, int lda,
                       const _Float16* __restrict__ Bs,
                       const float* __restrict__ bias,
                       const float* __restrict__ resid,
                       float* __restrict__ C, int ldc, int Kd, int Nn)
{
  int wave = threadIdx.x >> 5, lane = threadIdx.x & 31;
  int m0 = blockIdx.x * 16;
  int n0 = blockIdx.y * 128 + wave * 16;
  if (n0 >= Nn) return;                      // wave-uniform
  v8f acc = {};
  for (int k0 = 0; k0 < Kd; k0 += 32){
    v16h a = load_a_frag(A, lda, m0, k0, lane);
    v16h b = load_b_swz(Bs, Kd, k0, n0, lane);
    acc = WMMA_F32(a, b, acc);
  }
  int n = n0 + (lane & 15);
  int rb = (lane >> 4) * 8;
#pragma unroll
  for (int r = 0; r < 8; ++r){
    int m = m0 + rb + r;
    float v = acc[r];
    if (bias)  v += bias[n];
    if (ACT == 1) v = ssp_f(v);
    if (ACT == 2) v = silu_f(v);
    if (resid) v += resid[(size_t)m*ldc + n];
    C[(size_t)m*ldc + n] = v;
  }
}

// ---------------- fused edge filter: W_real = (ssp(ea@w1+b1)@w2 + b2) * C ----------------
// grid.x = E/16 ; block 256
__global__ void k_filt_wreal(const _Float16* __restrict__ ea_h,
                             const _Float16* __restrict__ w1s, const float* __restrict__ b1,
                             const _Float16* __restrict__ w2s, const float* __restrict__ b2,
                             const float* __restrict__ Cc, float* __restrict__ Wr)
{
  __shared__ __align__(32) _Float16 T[16*cF];
  int e0 = blockIdx.x * 16;
  int wave = threadIdx.x >> 5, lane = threadIdx.x & 31, n0 = wave * 16;
  v8f acc = {};
#pragma unroll
  for (int k0 = 0; k0 < cNGP; k0 += 32)
    acc = WMMA_F32(load_a_frag(ea_h, cNGP, e0, k0, lane), load_b_swz(w1s, cNGP, k0, n0, lane), acc);
  int n = n0 + (lane & 15), rb = (lane >> 4) * 8;
#pragma unroll
  for (int r = 0; r < 8; ++r) T[(rb+r)*cF + n] = (_Float16)ssp_f(acc[r] + b1[n]);
  __syncthreads();
  v8f acc2 = {};
#pragma unroll
  for (int k0 = 0; k0 < cF; k0 += 32)
    acc2 = WMMA_F32(load_a_frag(T, cF, 0, k0, lane), load_b_swz(w2s, cF, k0, n0, lane), acc2);
#pragma unroll
  for (int r = 0; r < 8; ++r){
    int row = rb + r;
    Wr[(size_t)(e0+row)*cF + n] = (acc2[r] + b2[n]) * Cc[e0+row];
  }
}

// ---------------- mask-score MLP (per node) ----------------
__global__ void k_ms_logit(const float* h, const float* w1, const float* b1,
                           const float* w2, const float* b2, float* logit){
  int v = blockIdx.x*blockDim.x + threadIdx.x; if (v >= cN) return;
  float acc = b2[0];
  for (int o = 0; o < cMSH; ++o){
    float s = b1[o];
    for (int k = 0; k < cSD; ++k) s += h[v*cH + k] * w1[k*cMSH + o];
    acc += silu_f(s) * w2[o];
  }
  logit[v] = acc;
}

// ---------------- top-K via single-block bitonic sort (descending) ----------------
__global__ void k_topk(const float* logit, int* midx, int* node2m){
  __shared__ float val[cN];
  __shared__ int   idx[cN];
  int t = threadIdx.x; // 1024 threads
  for (int i = t; i < cN; i += 1024){ val[i] = logit[i]; idx[i] = i; node2m[i] = -1; }
  __syncthreads();
  for (int k = 2; k <= cN; k <<= 1)
    for (int j = k >> 1; j > 0; j >>= 1){
      for (int i = t; i < cN; i += 1024){
        int ixj = i ^ j;
        if (ixj > i){
          bool up = ((i & k) == 0);                // "up" blocks descending
          float vi = val[i], vj = val[ixj];
          bool sw = up ? (vi < vj) : (vi > vj);
          if (sw){ val[i]=vj; val[ixj]=vi; int ti=idx[i]; idx[i]=idx[ixj]; idx[ixj]=ti; }
        }
      }
      __syncthreads();
    }
  if (t < cK){ midx[t] = idx[t]; node2m[idx[t]] = t; }
}

__global__ void k_gather_m(const float* h, const float* pos, const int* midx,
                           float* hm, _Float16* hm16, float* posm){
  int t = blockIdx.x*blockDim.x + threadIdx.x; if (t >= cK*cH) return;
  int m = t >> 7, c = t & 127; int v = midx[m];
  float x = h[v*cH + c]; hm[t] = x; hm16[t] = (_Float16)x;
  if (c < 3) posm[m*3 + c] = pos[v*3 + c];
}

__global__ void k_qk(const float* hm, const float* wq, const float* wk, float* q, float* kkv){
  int t = blockIdx.x*blockDim.x + threadIdx.x; if (t >= cK*cSD) return;
  int m = t >> 5, o = t & 31;
  float sq = 0.f, sk = 0.f;
  for (int k = 0; k < cSD; ++k){ float hv = hm[m*cH + k]; sq += hv*wq[k*cSD+o]; sk += hv*wk[k*cSD+o]; }
  q[t] = sq; kkv[t] = sk;
}

__global__ void k_adj_set(const int* ei, const int* node2m, float* adj){
  int e = blockIdx.x*blockDim.x + threadIdx.x; if (e >= cE) return;
  int mr = node2m[ei[e]], mc = node2m[ei[cE + e]];
  if (mr >= 0 && mc >= 0) adj[mr*cK + mc] = 1.f;
}

__global__ void k_av(const float* q, const float* kkv, const float* adj, float* Av){
  int t = blockIdx.x*blockDim.x + threadIdx.x; if (t >= cK*cK) return;
  int i = t / cK, j = t - i*cK;
  float lij = 0.f, lji = 0.f;
  for (int k = 0; k < cSD; ++k){
    lij += q[i*cSD+k]*kkv[j*cSD+k];
    lji += q[j*cSD+k]*kkv[i*cSD+k];
  }
  bool a1 = (lij > 0.f) && (adj[i*cK+j] == 0.f) && (i != j);   // sigmoid(0.1*lv)>0.5 <=> lv>0
  bool a2 = (lji > 0.f) && (adj[j*cK+i] == 0.f) && (i != j);
  Av[t] = (a1 || a2) ? 1.f : 0.f;
}

// ---------------- attention score (fused gv@W + silu + dot-w2) ----------------
// grid (K, K/16); block 256
__global__ void k_score(const float* __restrict__ posm,
                        const float* __restrict__ ahi, const float* __restrict__ ahj,
                        const _Float16* __restrict__ w1cs, const float* __restrict__ b1,
                        const float* __restrict__ w2v, const float* __restrict__ b2,
                        const float* __restrict__ Av, float* __restrict__ score)
{
  __shared__ __align__(32) _Float16 Ash[16*cNGP];
  __shared__ float red[16];
  int i = blockIdx.x, j0 = blockIdx.y * 16;
  int tid = threadIdx.x, wave = tid >> 5, lane = tid & 31;
  { // stage gv rows (16 j's for this i), padded to 64
    int rrow = tid >> 4, cb = (tid & 15) * 4;
    int j = j0 + rrow;
    float dx = posm[i*3]-posm[j*3], dy = posm[i*3+1]-posm[j*3+1], dz = posm[i*3+2]-posm[j*3+2];
    float dv = sqrtf(dx*dx + dy*dy + dz*dz + 1e-9f);
    for (int c = cb; c < cb+4; ++c){
      float v = 0.f;
      if (c < cNG){ float tt = dv - GSTEP*c; v = __expf(GCOEFF*tt*tt); }
      Ash[rrow*cNGP + c] = (_Float16)v;
    }
    if (tid < 16) red[tid] = 0.f;
  }
  __syncthreads();
  int n0 = wave * 16;
  v8f acc = {};
#pragma unroll
  for (int k0 = 0; k0 < cNGP; k0 += 32)
    acc = WMMA_F32(load_a_frag(Ash, cNGP, 0, k0, lane), load_b_swz(w1cs, cNGP, k0, n0, lane), acc);
  int n = n0 + (lane & 15), rb = (lane >> 4) * 8;
  float s[8];
#pragma unroll
  for (int r = 0; r < 8; ++r){
    int j = j0 + rb + r;
    float pre = acc[r] + ahi[i*cH + n] + ahj[j*cH + n] + b1[n];
    s[r] = silu_f(pre) * w2v[n];
  }
#pragma unroll
  for (int w = 1; w < 16; w <<= 1){
#pragma unroll
    for (int r = 0; r < 8; ++r) s[r] += __shfl_xor(s[r], w, 32);
  }
  if ((lane & 15) == 0){
#pragma unroll
    for (int r = 0; r < 8; ++r) atomicAdd(&red[rb + r], s[r]);
  }
  __syncthreads();
  if (tid < 16){
    int j = j0 + tid;
    float sc = red[tid] + b2[0];
    score[i*cK + j] = (Av[i*cK + j] > 0.f) ? sc : -1e9f;
  }
}

__global__ void k_softmax(const float* score, const float* Av, float* decay){
  __shared__ float sh[256];
  int i = blockIdx.x, t = threadIdx.x;
  float mx = -3.4e38f;
  for (int j = t; j < cK; j += 256) mx = fmaxf(mx, score[i*cK + j]);
  sh[t] = mx; __syncthreads();
  for (int s = 128; s > 0; s >>= 1){ if (t < s) sh[t] = fmaxf(sh[t], sh[t+s]); __syncthreads(); }
  mx = sh[0]; __syncthreads();
  float sm = 0.f;
  for (int j = t; j < cK; j += 256) sm += __expf(score[i*cK + j] - mx);
  sh[t] = sm; __syncthreads();
  for (int s = 128; s > 0; s >>= 1){ if (t < s) sh[t] += sh[t+s]; __syncthreads(); }
  float tot = sh[0];
  for (int j = t; j < cK; j += 256){
    float dv = __expf(score[i*cK + j] - mx) / tot;
    decay[i*cK + j] = (Av[i*cK + j] > 0.f) ? dv : 0.f;
  }
}

// ---------------- edge-scatter aggregations ----------------
__global__ void k_scatter_hier(const int* ei, const int* node2m, const float* x1m,
                               const float* Wr, float* agg){
  int t = blockIdx.x*blockDim.x + threadIdx.x; if (t >= cE*cF) return;
  int e = t >> 7, c = t & 127;
  int mr = node2m[ei[e]], mc = node2m[ei[cE + e]];
  if (mr >= 0 && mc >= 0) atomicAdd(&agg[mc*cF + c], x1m[mr*cF + c] * Wr[t]);
}
__global__ void k_scatter_full(const int* ei, const float* x1, const float* Wr, float* agg){
  int t = blockIdx.x*blockDim.x + threadIdx.x; if (t >= cE*cF) return;
  int e = t >> 7, c = t & 127;
  atomicAdd(&agg[ei[cE + e]*cF + c], x1[ei[e]*cF + c] * Wr[t]);
}

// ---------------- hierarchical einsum: agg[j] += sum_i x1m[i]*filt(gv_ij*decay_ij)*Cv*Av ----------------
// grid (K, K/16); block 256. Two fused WMMA GEMMs + in-register row reduction.
__global__ void k_hier(const float* __restrict__ posm, const float* __restrict__ decay,
                       const float* __restrict__ Av, const float* __restrict__ x1m,
                       const _Float16* __restrict__ w1s, const float* __restrict__ b1,
                       const _Float16* __restrict__ w2s, const float* __restrict__ b2,
                       float* __restrict__ agg)
{
  __shared__ __align__(32) _Float16 Ash[16*cNGP];
  __shared__ __align__(32) _Float16 T[16*cF];
  __shared__ float avs[16], cvs[16];
  __shared__ int flag;
  int j = blockIdx.x, i0 = blockIdx.y * 16;
  int tid = threadIdx.x, wave = tid >> 5, lane = tid & 31;
  if (tid == 0) flag = 0;
  __syncthreads();
  { // stage A rows = (gv_ij * decay_ij), 16 i's for this j
    int rrow = tid >> 4, cb = (tid & 15) * 4;
    int i = i0 + rrow;
    float dx = posm[i*3]-posm[j*3], dy = posm[i*3+1]-posm[j*3+1], dz = posm[i*3+2]-posm[j*3+2];
    float dv = sqrtf(dx*dx + dy*dy + dz*dz + 1e-9f);
    float dec = decay[i*cK + j];
    if ((tid & 15) == 0){
      float av = Av[i*cK + j];
      avs[rrow] = av;
      cvs[rrow] = 0.5f * (__cosf(dv * (PI_F/10.f)) + 1.f);
      if (av > 0.f) flag = 1;
    }
    for (int c = cb; c < cb+4; ++c){
      float v = 0.f;
      if (c < cNG){ float tt = dv - GSTEP*c; v = __expf(GCOEFF*tt*tt) * dec; }
      Ash[rrow*cNGP + c] = (_Float16)v;
    }
  }
  __syncthreads();
  if (flag == 0) return;                     // block-uniform
  int n0 = wave * 16;
  v8f acc = {};
#pragma unroll
  for (int k0 = 0; k0 < cNGP; k0 += 32)
    acc = WMMA_F32(load_a_frag(Ash, cNGP, 0, k0, lane), load_b_swz(w1s, cNGP, k0, n0, lane), acc);
  int n = n0 + (lane & 15), rb = (lane >> 4) * 8;
#pragma unroll
  for (int r = 0; r < 8; ++r) T[(rb+r)*cF + n] = (_Float16)ssp_f(acc[r] + b1[n]);
  __syncthreads();
  v8f acc2 = {};
#pragma unroll
  for (int k0 = 0; k0 < cF; k0 += 32)
    acc2 = WMMA_F32(load_a_frag(T, cF, 0, k0, lane), load_b_swz(w2s, cF, k0, n0, lane), acc2);
  float tot = 0.f;
#pragma unroll
  for (int r = 0; r < 8; ++r){
    int row = rb + r;
    float wv = (acc2[r] + b2[n]) * cvs[row];
    tot += wv * avs[row] * x1m[(i0 + row)*cF + n];
  }
  tot += __shfl_xor(tot, 16, 32);            // fold rows 8..15 onto 0..7 half (same column)
  if (lane < 16) atomicAdd(&agg[j*cF + n], tot);
}

// ---------------- gate merge + readout ----------------
__global__ void k_hfinal(const float* h, const float* hhier, const int* node2m,
                         const float* logit, float* hfin){
  int t = blockIdx.x*blockDim.x + threadIdx.x; if (t >= cN*cH) return;
  int v = t >> 7, c = t & 127;
  int jm = node2m[v];
  float hv = h[t];
  if (jm >= 0){
    float mg = sigm_f(logit[v]);             // TAU = 1
    hfin[t] = (1.f - mg)*hv + mg*hhier[jm*cH + c];
  } else hfin[t] = hv;
}

__global__ void k_finalize(const float* h2, const float* enw, const float* enb,
                           const float* chw, const float* chb, const int* batch, float* out){
  int v = blockIdx.x*blockDim.x + threadIdx.x; if (v >= cN) return;
  float e = enb[0], q = chb[0];
  for (int k = 0; k < cMSH; ++k){ float x = h2[v*cMSH + k]; e += x*enw[k]; q += x*chw[k]; }
  out[v] = e; out[cN + v] = q;
  atomicAdd(&out[2*cN + batch[v]], e);
  atomicAdd(&out[2*cN + cG + batch[v]], q);
}

// =====================================================================================
extern "C" void kernel_launch(void* const* d_in, const int* in_sizes, int n_in,
                              void* d_out, int out_size, void* d_ws, size_t ws_size,
                              hipStream_t stream)
{
  (void)in_sizes; (void)n_in; (void)out_size; (void)ws_size;
  const int*   atoms   = (const int*)  d_in[0];
  const float* pos     = (const float*)d_in[1];
  const int*   ei      = (const int*)  d_in[2];
  const int*   batch   = (const int*)  d_in[3];
  const float* emb     = (const float*)d_in[4];
  const float* lin1_w  = (const float*)d_in[5];
  const float* lin2_w  = (const float*)d_in[6];
  const float* lin2_b  = (const float*)d_in[7];
  const float* mlp_w1  = (const float*)d_in[8];
  const float* mlp_b1  = (const float*)d_in[9];
  const float* mlp_w2  = (const float*)d_in[10];
  const float* mlp_b2  = (const float*)d_in[11];
  const float* lin_w   = (const float*)d_in[12];
  const float* lin_b   = (const float*)d_in[13];
  const float* attn_w1 = (const float*)d_in[14];
  const float* attn_b1 = (const float*)d_in[15];
  const float* attn_w2 = (const float*)d_in[16];
  const float* attn_b2 = (const float*)d_in[17];
  const float* ms_w1   = (const float*)d_in[18];
  const float* ms_b1   = (const float*)d_in[19];
  const float* ms_w2   = (const float*)d_in[20];
  const float* ms_b2   = (const float*)d_in[21];
  const float* vg_wq   = (const float*)d_in[22];
  const float* vg_wk   = (const float*)d_in[23];
  const float* trunk_w = (const float*)d_in[24];
  const float* trunk_b = (const float*)d_in[25];
  const float* en_w    = (const float*)d_in[26];
  const float* en_b    = (const float*)d_in[27];
  const float* ch_w    = (const float*)d_in[28];
  const float* ch_b    = (const float*)d_in[29];
  float* out = (float*)d_out;

  // ---- workspace carve-up ----
  char* wsb = (char*)d_ws; size_t off = 0;
  auto alloc = [&](size_t bytes)->char*{ char* p = wsb + off; off += (bytes + 255) & ~(size_t)255; return p; };

  _Float16* ea_h    = (_Float16*)alloc((size_t)cE*cNGP*2);
  float*    Cc      = (float*)   alloc((size_t)cE*4);
  float*    Wreal   = (float*)   alloc((size_t)cE*cF*4);
  _Float16* s_mlp1  = (_Float16*)alloc((size_t)cL*cNGP*cF*2);
  _Float16* s_mlp2  = (_Float16*)alloc((size_t)cL*cF*cF*2);
  _Float16* s_lin1  = (_Float16*)alloc((size_t)cL*cH*cF*2);
  _Float16* s_lin2  = (_Float16*)alloc((size_t)cL*cF*cH*2);
  _Float16* s_lin   = (_Float16*)alloc((size_t)cL*cH*cH*2);
  _Float16* s_attna = (_Float16*)alloc((size_t)cL*cH*cH*2);
  _Float16* s_attnb = (_Float16*)alloc((size_t)cL*cH*cH*2);
  _Float16* s_attnc = (_Float16*)alloc((size_t)cL*cNGP*cH*2);
  _Float16* s_trunk = (_Float16*)alloc((size_t)cH*cMSH*2);
  float*    h       = (float*)   alloc((size_t)cN*cH*4);
  _Float16* h16     = (_Float16*)alloc((size_t)cN*cH*2);
  float*    hfin    = (float*)   alloc((size_t)cN*cH*4);
  _Float16* hf16    = (_Float16*)alloc((size_t)cN*cH*2);
  float*    logit   = (float*)   alloc((size_t)cN*4);
  int*      midx    = (int*)     alloc((size_t)cK*4);
  int*      node2m  = (int*)     alloc((size_t)cN*4);
  float*    hm      = (float*)   alloc((size_t)cK*cH*4);
  _Float16* hm16    = (_Float16*)alloc((size_t)cK*cH*2);
  float*    posm    = (float*)   alloc((size_t)cK*3*4);
  float*    qm      = (float*)   alloc((size_t)cK*cSD*4);
  float*    km      = (float*)   alloc((size_t)cK*cSD*4);
  float*    adj     = (float*)   alloc((size_t)cK*cK*4);
  float*    Avm     = (float*)   alloc((size_t)cK*cK*4);
  float*    score   = (float*)   alloc((size_t)cK*cK*4);
  float*    decay   = (float*)   alloc((size_t)cK*cK*4);
  float*    ahi     = (float*)   alloc((size_t)cK*cH*4);
  float*    ahj     = (float*)   alloc((size_t)cK*cH*4);
  float*    x1m     = (float*)   alloc((size_t)cK*cF*4);
  float*    aggK    = (float*)   alloc((size_t)cK*cF*4);
  _Float16* aggK16  = (_Float16*)alloc((size_t)cK*cF*2);
  float*    tmpK    = (float*)   alloc((size_t)cK*cH*4);
  _Float16* tmpK16  = (_Float16*)alloc((size_t)cK*cH*2);
  float*    hhier   = (float*)   alloc((size_t)cK*cH*4);
  float*    x1      = (float*)   alloc((size_t)cN*cF*4);
  float*    aggF    = (float*)   alloc((size_t)cN*cF*4);
  _Float16* aggF16  = (_Float16*)alloc((size_t)cN*cF*2);
  float*    tmpN    = (float*)   alloc((size_t)cN*cH*4);
  _Float16* tmpN16  = (_Float16*)alloc((size_t)cN*cH*2);
  float*    h2      = (float*)   alloc((size_t)cN*cMSH*4);

  auto nb = [](int n){ return (n + 255) / 256; };
  dim3 B(256);

  // ---- weight conversions: f32 -> swizzled fragment-major f16 (K padded 50->64 where needed) ----
  auto swz = [&](const float* src, int ldb, int roff, int kvalid, int Kd, int Nn, _Float16* dst){
    k_swzB<<<nb(Kd*Nn), B, 0, stream>>>(src, ldb, roff, kvalid, Kd, Nn, dst);
  };
  for (int l = 0; l < cL; ++l){
    swz(mlp_w1 + (size_t)l*cNG*cF,  cF, 0,    cNG, cNGP, cF, s_mlp1 + (size_t)l*cNGP*cF);
    swz(mlp_w2 + (size_t)l*cF*cF,   cF, 0,    cF,  cF,   cF, s_mlp2 + (size_t)l*cF*cF);
    swz(lin1_w + (size_t)l*cH*cF,   cF, 0,    cH,  cH,   cF, s_lin1 + (size_t)l*cH*cF);
    swz(lin2_w + (size_t)l*cF*cH,   cH, 0,    cF,  cF,   cH, s_lin2 + (size_t)l*cF*cH);
    swz(lin_w  + (size_t)l*cH*cH,   cH, 0,    cH,  cH,   cH, s_lin  + (size_t)l*cH*cH);
    swz(attn_w1 + (size_t)l*306*cH, cH, 0,    cH,  cH,   cH, s_attna + (size_t)l*cH*cH);
    swz(attn_w1 + (size_t)l*306*cH, cH, cH,   cH,  cH,   cH, s_attnb + (size_t)l*cH*cH);
    swz(attn_w1 + (size_t)l*306*cH, cH, 2*cH, cNG, cNGP, cH, s_attnc + (size_t)l*cNGP*cH);
  }
  swz(trunk_w, cMSH, 0, cH, cH, cMSH, s_trunk);

  // ---- graph setup ----
  k_embed<<<nb(cN*cH), B, 0, stream>>>(atoms, emb, h);
  k_f2h<<<nb(cN*cH), B, 0, stream>>>(h, h16, cN*cH);
  k_edge_geom<<<nb(cE), B, 0, stream>>>(ei, pos, ea_h, Cc);

  for (int l = 0; l < cL; ++l){
    const _Float16* wm1 = s_mlp1 + (size_t)l*cNGP*cF;
    const _Float16* wm2 = s_mlp2 + (size_t)l*cF*cF;
    const float* bm1 = mlp_b1 + l*cF;
    const float* bm2 = mlp_b2 + l*cF;

    // W_real over all edges (fused 2x WMMA GEMM)
    k_filt_wreal<<<dim3(cE/16), B, 0, stream>>>(ea_h, wm1, bm1, wm2, bm2, Cc, Wreal);

    // mask logits, top-K, gather
    k_ms_logit<<<nb(cN), B, 0, stream>>>(h, ms_w1 + l*cSD*cMSH, ms_b1 + l*cMSH,
                                         ms_w2 + l*cMSH, ms_b2 + l, logit);
    k_topk<<<1, 1024, 0, stream>>>(logit, midx, node2m);
    k_gather_m<<<nb(cK*cH), B, 0, stream>>>(h, pos, midx, hm, hm16, posm);
    k_qk<<<nb(cK*cSD), B, 0, stream>>>(hm, vg_wq + l*cSD*cSD, vg_wk + l*cSD*cSD, qm, km);
    k_zero<<<nb(cK*cK), B, 0, stream>>>(adj, cK*cK);
    k_adj_set<<<nb(cE), B, 0, stream>>>(ei, node2m, adj);
    k_av<<<nb(cK*cK), B, 0, stream>>>(qm, km, adj, Avm);

    // attention: a_hi / a_hj projections (WMMA GEMM) then fused score + softmax
    k_gemm<0><<<dim3(cK/16, 1), B, 0, stream>>>(hm16, cH, s_attna + (size_t)l*cH*cH,
                                                nullptr, nullptr, ahi, cH, cH, cH);
    k_gemm<0><<<dim3(cK/16, 1), B, 0, stream>>>(hm16, cH, s_attnb + (size_t)l*cH*cH,
                                                nullptr, nullptr, ahj, cH, cH, cH);
    k_score<<<dim3(cK, cK/16), B, 0, stream>>>(posm, ahi, ahj, s_attnc + (size_t)l*cNGP*cH,
                                               attn_b1 + l*cH, attn_w2 + l*cH, attn_b2 + l,
                                               Avm, score);
    k_softmax<<<dim3(cK), B, 0, stream>>>(score, Avm, decay);

    // x1m = h_m @ lin1_w
    k_gemm<0><<<dim3(cK/16, 1), B, 0, stream>>>(hm16, cH, s_lin1 + (size_t)l*cH*cF,
                                                nullptr, nullptr, x1m, cF, cH, cF);

    // agg = scatter(real edges among selected) + einsum over virtual graph
    k_zero<<<nb(cK*cF), B, 0, stream>>>(aggK, cK*cF);
    k_scatter_hier<<<nb(cE*cF), B, 0, stream>>>(ei, node2m, x1m, Wreal, aggK);
    k_hier<<<dim3(cK, cK/16), B, 0, stream>>>(posm, decay, Avm, x1m, wm1, bm1, wm2, bm2, aggK);

    // h_hier = h_m + post(agg)
    k_f2h<<<nb(cK*cF), B, 0, stream>>>(aggK, aggK16, cK*cF);
    k_gemm<1><<<dim3(cK/16, 1), B, 0, stream>>>(aggK16, cF, s_lin2 + (size_t)l*cF*cH,
                                                lin2_b + l*cH, nullptr, tmpK, cH, cF, cH);
    k_f2h<<<nb(cK*cH), B, 0, stream>>>(tmpK, tmpK16, cK*cH);
    k_gemm<0><<<dim3(cK/16, 1), B, 0, stream>>>(tmpK16, cH, s_lin + (size_t)l*cH*cH,
                                                lin_b + l*cH, hm, hhier, cH, cH, cH);

    // gate merge, then full-graph message passing
    k_hfinal<<<nb(cN*cH), B, 0, stream>>>(h, hhier, node2m, logit, hfin);
    k_f2h<<<nb(cN*cH), B, 0, stream>>>(hfin, hf16, cN*cH);
    k_gemm<0><<<dim3(cN/16, 1), B, 0, stream>>>(hf16, cH, s_lin1 + (size_t)l*cH*cF,
                                                nullptr, nullptr, x1, cF, cH, cF);
    k_zero<<<nb(cN*cF), B, 0, stream>>>(aggF, cN*cF);
    k_scatter_full<<<nb(cE*cF), B, 0, stream>>>(ei, x1, Wreal, aggF);
    k_f2h<<<nb(cN*cF), B, 0, stream>>>(aggF, aggF16, cN*cF);
    k_gemm<1><<<dim3(cN/16, 1), B, 0, stream>>>(aggF16, cF, s_lin2 + (size_t)l*cF*cH,
                                                lin2_b + l*cH, nullptr, tmpN, cH, cF, cH);
    k_f2h<<<nb(cN*cH), B, 0, stream>>>(tmpN, tmpN16, cN*cH);
    k_gemm<0><<<dim3(cN/16, 1), B, 0, stream>>>(tmpN16, cH, s_lin + (size_t)l*cH*cH,
                                                lin_b + l*cH, hfin, h, cH, cH, cH);
    k_f2h<<<nb(cN*cH), B, 0, stream>>>(h, h16, cN*cH);
  }

  // ---- readout ----
  k_gemm<2><<<dim3(cN/16, 1), B, 0, stream>>>(h16, cH, s_trunk, trunk_b, nullptr,
                                              h2, cMSH, cH, cMSH);
  k_zero<<<nb(2*cN + 2*cG), B, 0, stream>>>(out, 2*cN + 2*cG);
  k_finalize<<<nb(cN), B, 0, stream>>>(h2, en_w, en_b, ch_w, ch_b, batch, out);
}
